// StrideMoEOCR_77378130805210
// MI455X (gfx1250) — matrix operand order repaired and checked by reading, not compiled, not run on hardware
//
#include <hip/hip_runtime.h>
#include <hip/hip_bf16.h>
#include <cstdint>

// ---------------- problem constants ----------------
constexpr int NTOK = 2048;   // B*S
constexpr int DD   = 512;
constexpr int HH   = 2048;
constexpr int EE   = 8;
constexpr float LN_EPS = 1e-5f;

typedef __bf16 bf16_t;
typedef __attribute__((ext_vector_type(16))) __bf16 v16bf;
typedef __attribute__((ext_vector_type(8)))  __bf16 v8bf;
typedef __attribute__((ext_vector_type(4)))  __bf16 v4bf;
typedef __attribute__((ext_vector_type(8)))  float  v8f;

static __device__ __forceinline__ v16bf cat8(v8bf lo, v8bf hi) {
  return __builtin_shufflevector(lo, hi, 0,1,2,3,4,5,6,7,8,9,10,11,12,13,14,15);
}
// A-fragment: 16 contiguous bf16 at `off` within one LDS row
static __device__ __forceinline__ v16bf ldsA(const bf16_t* row, int off) {
  return cat8(*(const v8bf*)(row + off), *(const v8bf*)(row + off + 8));
}

// ---------------- kernel 0: zero out + counters/stats ----------------
__global__ void k_zero(float* out, int* counters, float* statAcc) {
  int idx = blockIdx.x * blockDim.x + threadIdx.x;
  if (idx < NTOK * DD) out[idx] = 0.0f;
  if (idx < EE) counters[idx] = 0;
  if (idx < 10) statAcc[idx] = 0.0f;
}

// ---------------- kernel 1: M2 = ctx_W @ rtr_W  (512 x 8) ----------------
__global__ void k_prep_m2(const float* __restrict__ ctx_W,
                          const float* __restrict__ rtr_W,
                          float* __restrict__ M2) {
  int idx = blockIdx.x * blockDim.x + threadIdx.x;   // 4096 threads
  if (idx >= DD * EE) return;
  int k = idx >> 3, e = idx & 7;
  float s = 0.0f;
  for (int d = 0; d < DD; ++d) s += ctx_W[(size_t)k * DD + d] * rtr_W[d * EE + e];
  M2[k * EE + e] = s;
}

// ---------------- kernel 2: vb, qv fold vectors ----------------
__global__ void k_prep_vec(const float* __restrict__ rtr_W, const float* __restrict__ rtr_b,
                           const float* __restrict__ ctx_b, const float* __restrict__ qual_b,
                           const float* __restrict__ rn_b,  const float* __restrict__ cn_b,
                           const float* __restrict__ qual_W, const float* __restrict__ M2,
                           float* __restrict__ vb, float* __restrict__ qv) {
  int t = threadIdx.x;
  if (t < EE) {
    int e = t;
    float s = rtr_b[e];
    for (int d = 0; d < DD; ++d) s += (ctx_b[d] + qual_b[d] + rn_b[d]) * rtr_W[d * EE + e];
    for (int d = 0; d < DD; ++d) s += cn_b[d] * M2[d * EE + e];
    vb[e] = s;
  } else if (t < 2 * EE) {
    int e = t - EE;
    float s = 0.0f;
    for (int d = 0; d < DD; ++d) s += qual_W[d] * rtr_W[d * EE + e];
    qv[e] = s;
  }
}

// ---------------- kernel 3: weight transpose + fp32 -> bf16 ----------------
// W1 (e,d,h) -> W1T (e,h,d) ; W2 (e,h,d) -> W2T (e,d,h)
__global__ void k_wt_cvt(const float* __restrict__ W1, const float* __restrict__ W2,
                         bf16_t* __restrict__ W1T, bf16_t* __restrict__ W2T) {
  __shared__ float tile[32][33];
  int zz = blockIdx.z;
  int which = zz >> 3, e = zz & 7;
  int R, C; const float* src; bf16_t* dst;
  if (which == 0) { R = DD; C = HH; src = W1 + (size_t)e * DD * HH; dst = W1T + (size_t)e * DD * HH; }
  else            { R = HH; C = DD; src = W2 + (size_t)e * DD * HH; dst = W2T + (size_t)e * DD * HH; }
  int x0 = blockIdx.x * 32, y0 = blockIdx.y * 32;
  if (x0 >= C || y0 >= R) return;
  int tx = threadIdx.x;
  for (int j = threadIdx.y; j < 32; j += 8)
    tile[j][tx] = src[(size_t)(y0 + j) * C + x0 + tx];
  __syncthreads();
  for (int j = threadIdx.y; j < 32; j += 8)
    dst[(size_t)(x0 + j) * R + y0 + tx] = (bf16_t)tile[tx][j];
}

// ---------------- kernel 4: router (one wave32 per token) ----------------
__global__ void __launch_bounds__(256)
k_router(const float* __restrict__ x, const float* __restrict__ ctx,
         const float* __restrict__ quality,
         const float* __restrict__ rn_g, const float* __restrict__ cn_g,
         const float* __restrict__ rtr_W, const float* __restrict__ M2,
         const float* __restrict__ vb, const float* __restrict__ qv,
         const float* __restrict__ temperature,
         int* __restrict__ counters, int* __restrict__ tokList,
         float* __restrict__ gateList, float* __restrict__ statAcc) {
  const int wv = threadIdx.x >> 5;
  const int lane = threadIdx.x & 31;
  const int t = blockIdx.x * 8 + wv;
  const float* xr = x + (size_t)t * DD;
  const float* cr = ctx + (size_t)t * DD;

  // pass 1: mean/var of both rows
  float sx = 0.f, sxx = 0.f, sc = 0.f, scc = 0.f;
  for (int k = lane; k < DD; k += 32) {
    float xv = xr[k]; sx += xv; sxx += xv * xv;
    float cv = cr[k]; sc += cv; scc += cv * cv;
  }
  for (int off = 16; off > 0; off >>= 1) {
    sx += __shfl_xor(sx, off); sxx += __shfl_xor(sxx, off);
    sc += __shfl_xor(sc, off); scc += __shfl_xor(scc, off);
  }
  float mx = sx * (1.0f / DD), vx = sxx * (1.0f / DD) - mx * mx;
  float mc = sc * (1.0f / DD), vc = scc * (1.0f / DD) - mc * mc;
  float rx = rsqrtf(vx + LN_EPS), rc = rsqrtf(vc + LN_EPS);

  // pass 2: fused dual dot products against rtr_W and M2 columns
  float ax[EE], ac[EE];
  #pragma unroll
  for (int e = 0; e < EE; ++e) { ax[e] = 0.f; ac[e] = 0.f; }
  for (int k = lane; k < DD; k += 32) {
    float xh = (xr[k] - mx) * rx * rn_g[k];
    float ch = (cr[k] - mc) * rc * cn_g[k];
    const float4* rw = (const float4*)(rtr_W + k * EE);
    float4 r0 = rw[0], r1 = rw[1];
    ax[0] += xh * r0.x; ax[1] += xh * r0.y; ax[2] += xh * r0.z; ax[3] += xh * r0.w;
    ax[4] += xh * r1.x; ax[5] += xh * r1.y; ax[6] += xh * r1.z; ax[7] += xh * r1.w;
    const float4* mw = (const float4*)(M2 + k * EE);
    float4 m0 = mw[0], m1 = mw[1];
    ac[0] += ch * m0.x; ac[1] += ch * m0.y; ac[2] += ch * m0.z; ac[3] += ch * m0.w;
    ac[4] += ch * m1.x; ac[5] += ch * m1.y; ac[6] += ch * m1.z; ac[7] += ch * m1.w;
  }
  #pragma unroll
  for (int e = 0; e < EE; ++e)
    for (int off = 16; off > 0; off >>= 1) {
      ax[e] += __shfl_xor(ax[e], off);
      ac[e] += __shfl_xor(ac[e], off);
    }

  if (lane == 0) {
    float q = quality[t >> 9];                     // batch = token / S
    float invT = 1.0f / fmaxf(temperature[0], 0.25f);
    float lg[EE];
    #pragma unroll
    for (int e = 0; e < EE; ++e) lg[e] = (ax[e] + ac[e] + vb[e] + q * qv[e]) * invT;
    // top-2 (first max wins ties, matching lax.top_k)
    int i0 = 0;
    #pragma unroll
    for (int e = 1; e < EE; ++e) if (lg[e] > lg[i0]) i0 = e;
    int i1 = (i0 == 0) ? 1 : 0;
    #pragma unroll
    for (int e = 0; e < EE; ++e) if (e != i0 && lg[e] > lg[i1]) i1 = e;
    float e1 = __expf(lg[i1] - lg[i0]);
    float g0 = 1.0f / (1.0f + e1), g1 = e1 * g0;
    int p0 = atomicAdd(&counters[i0], 1);
    tokList[i0 * NTOK + p0] = t; gateList[i0 * NTOK + p0] = g0;
    int p1 = atomicAdd(&counters[i1], 1);
    tokList[i1 * NTOK + p1] = t; gateList[i1 * NTOK + p1] = g1;
    // stats: softmax over all 8
    float mmax = lg[i0], se = 0.f, pe[EE];
    #pragma unroll
    for (int e = 0; e < EE; ++e) { pe[e] = __expf(lg[e] - mmax); se += pe[e]; }
    float lse = mmax + __logf(se), rse = 1.0f / se, ent = 0.f;
    #pragma unroll
    for (int e = 0; e < EE; ++e) {
      float p = pe[e] * rse;
      unsafeAtomicAdd(&statAcc[e], p);
      ent -= p * __logf(fmaxf(p, 1e-9f));
    }
    unsafeAtomicAdd(&statAcc[8], lse * lse);
    unsafeAtomicAdd(&statAcc[9], ent);
  }
}

// ---------------- kernel 5: grouped expert GEMM (bf16 WMMA) ----------------
// grid (E, 64 tiles of 32 tokens), block 256 = 8 waves.
// M-tile = 32 tokens, kb-outer / N-tile-inner: each A pair (4 ds_load_b128)
// feeds 8 WMMAs; each B fragment (2 global_load_b128) feeds 2 WMMAs.
__global__ void __launch_bounds__(256)
k_expert(const float* __restrict__ x,
         const bf16_t* __restrict__ W1T, const bf16_t* __restrict__ W2T,
         const float* __restrict__ b1, const float* __restrict__ b2,
         const int* __restrict__ counters, const int* __restrict__ tokList,
         const float* __restrict__ gateList, float* __restrict__ out) {
  constexpr int MT = 32;                            // tokens per block
  constexpr int PITCH = 520;                        // bf16 elems, 16B-aligned rows
  __shared__ bf16_t Xs[MT][PITCH];
  __shared__ bf16_t Hs[MT][PITCH];
  __shared__ int   sTok[MT];
  __shared__ float sGate[MT];

  const int e = blockIdx.x;
  const int count = counters[e];
  const int mBase = blockIdx.y * MT;
  if (mBase >= count) return;

  const int tid = threadIdx.x;
  if (tid < MT) {
    int row = mBase + tid;
    if (row < count) { sTok[tid] = tokList[e * NTOK + row]; sGate[tid] = gateList[e * NTOK + row]; }
    else             { sTok[tid] = 0;                       sGate[tid] = 0.0f; }
  }
  __syncthreads();

  // stage X tile (32 gathered token rows) as bf16 into LDS, float4 at a time
  for (int idx = tid; idx < MT * (DD / 4); idx += 256) {
    int r = idx >> 7, c4 = idx & (DD / 4 - 1);
    float4 v = ((const float4*)(x + (size_t)sTok[r] * DD))[c4];
    v4bf p = { (bf16_t)v.x, (bf16_t)v.y, (bf16_t)v.z, (bf16_t)v.w };
    *(v4bf*)&Xs[r][c4 * 4] = p;
  }
  __syncthreads();

  const int wv   = tid >> 5;
  const int lane = tid & 31;
  const int mA   = lane & 15;       // A-frag row (within M-subtile)
  const int kh   = lane >> 4;       // K-half selector
  const int nCol = lane & 15;       // B / C column
  const int rBase = kh * 8;         // C rows (within M-subtile)
  const int aOff  = kh * 16;        // A-frag K offset within row
  const int bOff  = kh * 16;        // B-frag K offset within column

  const bf16_t* W1e = W1T + (size_t)e * DD * HH;   // [h][d]
  const bf16_t* W2e = W2T + (size_t)e * DD * HH;   // [d][h]

  v8f yacc[4][2];
  #pragma unroll
  for (int i = 0; i < 4; ++i)
    #pragma unroll
    for (int m = 0; m < 2; ++m)
      yacc[i][m] = (v8f){0.f,0.f,0.f,0.f,0.f,0.f,0.f,0.f};

  for (int ch = 0; ch < HH / DD; ++ch) {           // 4 H-chunks of 512
    // ---- phase 1: Hs = gelu(X @ W1[:, chunk] + b1) ----
    {
      v8f acc[4][2];
      #pragma unroll
      for (int i = 0; i < 4; ++i) {
        acc[i][0] = (v8f){0.f,0.f,0.f,0.f,0.f,0.f,0.f,0.f};
        acc[i][1] = acc[i][0];
      }
      const bf16_t* bpBase = W1e + (size_t)(ch * DD + wv * 64 + nCol) * DD;
      for (int kb = 0; kb < DD; kb += 32) {
        v16bf a0 = ldsA(&Xs[mA][0],      kb + aOff);
        v16bf a1 = ldsA(&Xs[16 + mA][0], kb + aOff);
        #pragma unroll
        for (int i = 0; i < 4; ++i) {
          v16bf b = *(const v16bf*)(bpBase + (size_t)i * 16 * DD + kb + bOff);
          acc[i][0] = __builtin_amdgcn_wmma_f32_16x16x32_bf16(false, a0, false, b,
                                                              (short)0, acc[i][0], false, false);
          acc[i][1] = __builtin_amdgcn_wmma_f32_16x16x32_bf16(false, a1, false, b,
                                                              (short)0, acc[i][1], false, false);
        }
      }
      #pragma unroll
      for (int i = 0; i < 4; ++i) {
        int hLoc = wv * 64 + i * 16 + nCol;
        float bias = b1[e * HH + ch * DD + hLoc];
        #pragma unroll
        for (int r = 0; r < 8; ++r) {
          float v0 = acc[i][0][r] + bias;
          float v1 = acc[i][1][r] + bias;
          Hs[rBase + r][hLoc]      = (bf16_t)(0.5f * v0 * (1.0f + erff(v0 * 0.70710678118654752f)));
          Hs[16 + rBase + r][hLoc] = (bf16_t)(0.5f * v1 * (1.0f + erff(v1 * 0.70710678118654752f)));
        }
      }
    }
    __syncthreads();
    // ---- phase 2: Y += Hs @ W2[chunk, :] ----
    {
      const bf16_t* bpBase = W2e + (size_t)(wv * 64 + nCol) * HH + ch * DD;
      for (int kb = 0; kb < DD; kb += 32) {
        v16bf a0 = ldsA(&Hs[mA][0],      kb + aOff);
        v16bf a1 = ldsA(&Hs[16 + mA][0], kb + aOff);
        #pragma unroll
        for (int i = 0; i < 4; ++i) {
          v16bf b = *(const v16bf*)(bpBase + (size_t)i * 16 * HH + kb + bOff);
          yacc[i][0] = __builtin_amdgcn_wmma_f32_16x16x32_bf16(false, a0, false, b,
                                                               (short)0, yacc[i][0], false, false);
          yacc[i][1] = __builtin_amdgcn_wmma_f32_16x16x32_bf16(false, a1, false, b,
                                                               (short)0, yacc[i][1], false, false);
        }
      }
    }
    __syncthreads();                                // Hs reused next chunk
  }

  // ---- epilogue: out[token] += gate * (y + b2) ----
  #pragma unroll
  for (int i = 0; i < 4; ++i) {
    int dCol = wv * 64 + i * 16 + nCol;
    float bias = b2[e * DD + dCol];
    #pragma unroll
    for (int m = 0; m < 2; ++m) {
      #pragma unroll
      for (int r = 0; r < 8; ++r) {
        int mr = m * 16 + rBase + r;
        float g = sGate[mr];
        if (g != 0.0f)
          unsafeAtomicAdd(&out[(size_t)sTok[mr] * DD + dCol], g * (yacc[i][m][r] + bias));
      }
    }
  }
}

// ---------------- kernel 6: finalize scalars ----------------
__global__ void k_finalize(const float* __restrict__ statAcc, float* __restrict__ out) {
  if (threadIdx.x != 0) return;
  float lb = 0.0f;
  for (int e = 0; e < EE; ++e) {
    float d = statAcc[e] * (1.0f / NTOK) - 1.0f / EE;
    lb += d * d;
  }
  lb *= (1.0f / EE);
  float z = statAcc[8] * (1.0f / NTOK);
  float ent = statAcc[9] * (1.0f / NTOK);
  float* s = out + (size_t)NTOK * DD;
  s[0] = lb;
  s[1] = z;
  s[2] = ent;
  s[3] = lb + 0.001f * z - 0.001f * ent;
}

// ---------------- host launch ----------------
extern "C" void kernel_launch(void* const* d_in, const int* in_sizes, int n_in,
                              void* d_out, int out_size, void* d_ws, size_t ws_size,
                              hipStream_t stream) {
  const float* x       = (const float*)d_in[0];
  const float* ctx     = (const float*)d_in[1];
  const float* quality = (const float*)d_in[2];
  const float* rn_g    = (const float*)d_in[3];
  const float* rn_b    = (const float*)d_in[4];
  const float* cn_g    = (const float*)d_in[5];
  const float* cn_b    = (const float*)d_in[6];
  const float* ctx_W   = (const float*)d_in[7];
  const float* ctx_b   = (const float*)d_in[8];
  const float* qual_W  = (const float*)d_in[9];
  const float* qual_b  = (const float*)d_in[10];
  const float* rtr_W   = (const float*)d_in[11];
  const float* rtr_b   = (const float*)d_in[12];
  const float* temp    = (const float*)d_in[13];
  const float* W1      = (const float*)d_in[14];
  const float* b1      = (const float*)d_in[15];
  const float* W2      = (const float*)d_in[16];
  const float* b2      = (const float*)d_in[17];
  float* out = (float*)d_out;

  char* ws = (char*)d_ws;
  const size_t WELEMS = (size_t)EE * DD * HH;          // 8.4M
  bf16_t* W1T      = (bf16_t*)(ws);                    // 16 MB
  bf16_t* W2T      = (bf16_t*)(ws + WELEMS * 2);       // 16 MB
  size_t o = WELEMS * 4;
  float* M2        = (float*)(ws + o); o += DD * EE * 4;
  float* vb        = (float*)(ws + o); o += 32;
  float* qv        = (float*)(ws + o); o += 32;
  int*   counters  = (int*)  (ws + o); o += 32;
  int*   tokList   = (int*)  (ws + o); o += (size_t)EE * NTOK * 4;
  float* gateList  = (float*)(ws + o); o += (size_t)EE * NTOK * 4;
  float* statAcc   = (float*)(ws + o);

  k_zero<<<(NTOK * DD + 255) / 256, 256, 0, stream>>>(out, counters, statAcc);
  k_prep_m2<<<(DD * EE + 255) / 256, 256, 0, stream>>>(ctx_W, rtr_W, M2);
  k_prep_vec<<<1, 32, 0, stream>>>(rtr_W, rtr_b, ctx_b, qual_b, rn_b, cn_b, qual_W, M2, vb, qv);
  k_wt_cvt<<<dim3(64, 64, 16), dim3(32, 8), 0, stream>>>(W1, W2, W1T, W2T);
  k_router<<<NTOK / 8, 256, 0, stream>>>(x, ctx, quality, rn_g, cn_g, rtr_W, M2, vb, qv,
                                         temp, counters, tokList, gateList, statAcc);
  k_expert<<<dim3(EE, NTOK / 32), 256, 0, stream>>>(x, W1T, W2T, b1, b2,
                                                    counters, tokList, gateList, out);
  k_finalize<<<1, 32, 0, stream>>>(statAcc, out);
}
